// mambaLG_10256381903623
// MI455X (gfx1250) — matrix operand order repaired and checked by reading, not compiled, not run on hardware
//
#include <hip/hip_runtime.h>
#include <math.h>

// ---------------------------------------------------------------------------
// CDNA5 (gfx1250) MambaHSI-style forward pass.
// All channel contractions use v_wmma_f32_16x16x32_f16 (wave32 WMMA); each
// wave computes a 16x64 strip (4 WMMAs per A fragment) with fused epilogues.
// ---------------------------------------------------------------------------

typedef __attribute__((ext_vector_type(16))) _Float16 v16h;
typedef __attribute__((ext_vector_type(8)))  float    v8f;

static __device__ __forceinline__ float gelu_f(float x) {
  return 0.5f * x * (1.0f + erff(x * 0.70710678118654752440f));
}
static __device__ __forceinline__ float silu_f(float x) {
  return x / (1.0f + __expf(-x));
}

// epilogue modes
#define GM_NONE    0   // C = acc
#define GM_BIAS    1   // C = acc + bias[n]
#define GM_BNGELU  2   // C = gelu(bng[n]*(acc+bias[n])/sqrt(1+1e-5) + bnb[n])
#define GM_GELU    3   // C = gelu(acc + bias[n])
#define GM_ACC     4   // C += acc   (residual accumulate)

// ------------------------- generic WMMA GEMM -------------------------------
// C[M,N] (+)= A[M,K] * W[N,K]^T.  One wave per 16x64 C strip (4 subtiles).
__global__ void gemm_wmma_k(const float* __restrict__ A, const float* __restrict__ W,
                            const float* __restrict__ bias,
                            const float* __restrict__ bng, const float* __restrict__ bnb,
                            float* __restrict__ C, int M, int N, int K, int mode) {
  const int lane = threadIdx.x;                 // 0..31
  const int mt = blockIdx.x * blockDim.y + threadIdx.y;
  const int m0 = mt * 16;
  if (m0 >= M) return;                          // wave-uniform exit
  const int ntb = blockIdx.y * 4;               // first of 4 N-subtiles
  const int g = lane >> 4, r = lane & 15;
  v8f acc0 = {}, acc1 = {}, acc2 = {}, acc3 = {};

  for (int kb = 0; kb < K; kb += 32) {
    // ---- A fragment (shared by all 4 subtiles) ----
    v16h a;
    if ((m0 + 16 <= M) && (kb + 32 <= K)) {
      const float* ap = A + (size_t)(m0 + r) * K + kb + 16 * g;
      if (kb + 64 <= K) __builtin_prefetch(ap + 32, 0, 1);   // global_prefetch_b8
      float4 a0 = *reinterpret_cast<const float4*>(ap);
      float4 a1 = *reinterpret_cast<const float4*>(ap + 4);
      float4 a2 = *reinterpret_cast<const float4*>(ap + 8);
      float4 a3 = *reinterpret_cast<const float4*>(ap + 12);
      float av[16] = {a0.x,a0.y,a0.z,a0.w, a1.x,a1.y,a1.z,a1.w,
                      a2.x,a2.y,a2.z,a2.w, a3.x,a3.y,a3.z,a3.w};
#pragma unroll
      for (int e = 0; e < 16; ++e) a[e] = (_Float16)av[e];
    } else {
#pragma unroll
      for (int e = 0; e < 16; ++e) {
        int ka = kb + e + 16 * g;
        int m  = m0 + r;
        a[e] = (_Float16)((m < M && ka < K) ? A[(size_t)m * K + ka] : 0.0f);
      }
    }
    // ---- 4 B fragments / WMMAs ----
#pragma unroll
    for (int j = 0; j < 4; ++j) {
      const int n0 = (ntb + j) * 16;
      if (n0 >= N) continue;                    // wave-uniform
      v16h b;
      if ((n0 + 16 <= N) && (kb + 32 <= K)) {
        const float* bp = W + (size_t)(n0 + r) * K + kb + 16 * g;
        float4 b0 = *reinterpret_cast<const float4*>(bp);
        float4 b1 = *reinterpret_cast<const float4*>(bp + 4);
        float4 b2 = *reinterpret_cast<const float4*>(bp + 8);
        float4 b3 = *reinterpret_cast<const float4*>(bp + 12);
        float bv[16] = {b0.x,b0.y,b0.z,b0.w, b1.x,b1.y,b1.z,b1.w,
                        b2.x,b2.y,b2.z,b2.w, b3.x,b3.y,b3.z,b3.w};
#pragma unroll
        for (int e = 0; e < 16; ++e) b[e] = (_Float16)bv[e];
      } else {
#pragma unroll
        for (int e = 0; e < 16; ++e) {
          int kn = kb + e + 16 * g;
          int n  = n0 + r;
          b[e] = (_Float16)((n < N && kn < K) ? W[(size_t)n * K + kn] : 0.0f);
        }
      }
      v8f* accp = (j == 0) ? &acc0 : (j == 1) ? &acc1 : (j == 2) ? &acc2 : &acc3;
      *accp = __builtin_amdgcn_wmma_f32_16x16x32_f16(false, a, false, b,
                                                     (short)0, *accp, false, false);
    }
  }

  // ---- epilogue + store ----
  const float bninv = 0.99999500003749966f;     // 1/sqrt(1+1e-5)
#pragma unroll
  for (int j = 0; j < 4; ++j) {
    const int n0 = (ntb + j) * 16;
    if (n0 >= N) continue;
    v8f accv = (j == 0) ? acc0 : (j == 1) ? acc1 : (j == 2) ? acc2 : acc3;
#pragma unroll
    for (int i = 0; i < 8; ++i) {
      int m = m0 + i + 8 * g, n = n0 + r;
      if (m < M && n < N) {
        float v = accv[i];
        size_t idx = (size_t)m * N + n;
        if (mode == GM_BIAS)        v += bias[n];
        else if (mode == GM_BNGELU) v = gelu_f(bng[n] * (v + bias[n]) * bninv + bnb[n]);
        else if (mode == GM_GELU)   v = gelu_f(v + bias[n]);
        else if (mode == GM_ACC)    v += C[idx];
        C[idx] = v;
      }
    }
  }
}

// ------------------------- norm kernel -------------------------------------
__global__ void ln_rows_k(const float* __restrict__ in, float* __restrict__ out,
                          const float* __restrict__ g, const float* __restrict__ b,
                          int rows, int D, int do_gelu) {
  int r = blockIdx.x * blockDim.x + threadIdx.x;
  if (r >= rows) return;
  const float* xr = in + (size_t)r * D;
  float m = 0.0f;
  for (int i = 0; i < D; ++i) m += xr[i];
  m /= (float)D;
  float v = 0.0f;
  for (int i = 0; i < D; ++i) { float d = xr[i] - m; v += d * d; }
  v /= (float)D;
  float inv = rsqrtf(v + 1e-5f);
  float* orow = out + (size_t)r * D;
  for (int i = 0; i < D; ++i) {
    float y = (xr[i] - m) * inv * g[i] + b[i];
    orow[i] = do_gelu ? gelu_f(y) : y;
  }
}

// ------------------------- CNN branch kernels ------------------------------
__global__ void conv3_k(const float* __restrict__ x0, const float* __restrict__ w,
                        const float* __restrict__ bias,
                        const float* __restrict__ bg, const float* __restrict__ bb,
                        float* __restrict__ x2) {
  int i = blockIdx.x * blockDim.x + threadIdx.x;
  if (i >= 2 * 64 * 64 * 64) return;
  int d = i & 63, wp = (i >> 6) & 63, hp = (i >> 12) & 63, b = i >> 18;
  float acc = bias[d];
  for (int kh = 0; kh < 3; ++kh) {
    int h = hp + kh - 1; if (h < 0 || h >= 64) continue;
    for (int kw = 0; kw < 3; ++kw) {
      int ww = wp + kw - 1; if (ww < 0 || ww >= 64) continue;
      const float* xp = x0 + (((size_t)(b * 64 + h) * 64 + ww) * 64);
      const float* wq = w + ((size_t)(kh * 3 + kw) * 64) * 64 + d;
      for (int ci = 0; ci < 64; ++ci) acc += xp[ci] * wq[(size_t)ci * 64];
    }
  }
  x2[i] = gelu_f(bg[d] * acc * 0.99999500003749966f + bb[d]);
}

__global__ void pool5_k(const float* __restrict__ x0,
                        const float* __restrict__ bg, const float* __restrict__ bb,
                        float* __restrict__ x3) {
  int i = blockIdx.x * blockDim.x + threadIdx.x;
  if (i >= 2 * 64 * 64 * 64) return;
  int d = i & 63, wp = (i >> 6) & 63, hp = (i >> 12) & 63, b = i >> 18;
  float acc = 0.0f;
  for (int dh = -2; dh <= 2; ++dh) {
    int h = hp + dh; if (h < 0 || h >= 64) continue;
    for (int dw = -2; dw <= 2; ++dw) {
      int ww = wp + dw; if (ww < 0 || ww >= 64) continue;
      acc += x0[(((size_t)(b * 64 + h) * 64 + ww) * 64) + d];
    }
  }
  acc *= (1.0f / 25.0f);
  x3[i] = gelu_f(bg[d] * acc * 0.99999500003749966f + bb[d]);
}

__global__ void concat_k(const float* __restrict__ x0, const float* __restrict__ x1,
                         const float* __restrict__ x2, const float* __restrict__ x3,
                         float* __restrict__ xc) {
  int i = blockIdx.x * blockDim.x + threadIdx.x;
  if (i >= 8192 * 256) return;
  int row = i >> 8, c = i & 255, sel = c >> 6, cc = c & 63;
  const float* s = (sel == 0) ? x0 : (sel == 1) ? x1 : (sel == 2) ? x2 : x3;
  xc[i] = s[(size_t)row * 64 + cc];
}

// ------------------------- windowing (reflect pad 80x80, 16x16 tiles) ------
__global__ void window_gather_k(const float* __restrict__ xf, float* __restrict__ xw) {
  int i = blockIdx.x * blockDim.x + threadIdx.x;
  if (i >= 50 * 256 * 64) return;
  int c = i & 63, l = (i >> 6) & 255, win = i >> 14;
  int b = win / 25, wr = win % 25, nhi = wr / 5, nwi = wr % 5;
  int wt = l >> 4, ht = l & 15;
  int hh = nhi * 16 + ht, ww = nwi * 16 + wt;
  int sh = (hh < 64) ? hh : 126 - hh;          // reflect (no edge repeat)
  int sw = (ww < 64) ? ww : 126 - ww;
  xw[i] = xf[(((size_t)(b * 64 + sh) * 64 + sw) * 64) + c];
}

__global__ void window_scatter_add_k(const float* __restrict__ xw,
                                     const float* __restrict__ xf,
                                     float* __restrict__ xa) {
  int i = blockIdx.x * blockDim.x + threadIdx.x;
  if (i >= 8192 * 64) return;
  int c = i & 63, w = (i >> 6) & 63, h = (i >> 12) & 63, b = i >> 18;
  int nhi = h >> 4, ht = h & 15, nwi = w >> 4, wt = w & 15;
  int win = (b * 5 + nhi) * 5 + nwi;
  int l = wt * 16 + ht;
  xa[i] = xw[(((size_t)win * 256 + l) * 64) + c] + xf[i];
}

// ------------------------- Mamba kernels -----------------------------------
__global__ void dwconv_silu_k(const float* __restrict__ xz, const float* __restrict__ cw,
                              const float* __restrict__ cb, float* __restrict__ uc,
                              int Nseq, int L, int di) {
  int i = blockIdx.x * blockDim.x + threadIdx.x;
  if (i >= Nseq * L * di) return;
  int c = i % di, l = (i / di) % L, seq = i / (di * L);
  float acc = cb[c];
#pragma unroll
  for (int j = 0; j < 4; ++j) {
    int ll = l - 3 + j;
    if (ll >= 0) acc += xz[((size_t)(seq * L + ll) * 2 * di) + c] * cw[c * 4 + j];
  }
  uc[(size_t)(seq * L + l) * di + c] = silu_f(acc);
}

__global__ void delta_k(const float* __restrict__ dbl, const float* __restrict__ dtw,
                        const float* __restrict__ dtb, float* __restrict__ dlt,
                        int NL, int di, int dtr) {
  int i = blockIdx.x * blockDim.x + threadIdx.x;
  if (i >= NL * di) return;
  int c = i % di, row = i / di;
  int nx = dtr + 128;
  float acc = dtb[c];
  for (int r = 0; r < dtr; ++r) acc += dbl[(size_t)row * nx + r] * dtw[c * dtr + r];
  dlt[i] = (acc > 20.0f) ? acc : log1pf(expf(acc));     // softplus
}

// One 128-thread block per sequence; h-state (64 f32) and -exp(A_log) row
// kept fully in VGPRs; B/C vectors staged through LDS each step.
__global__ void scan_k(const float* __restrict__ dbl, const float* __restrict__ dlt,
                       const float* __restrict__ uc, const float* __restrict__ xz,
                       const float* __restrict__ Alog, const float* __restrict__ Dp,
                       float* __restrict__ yv, int L, int di, int dtr) {
  __shared__ float Bs[64];
  __shared__ float Cs[64];
  const int seq = blockIdx.x;
  const int t = threadIdx.x;                   // blockDim.x == 128
  const bool active = (t < di);
  const int nx = dtr + 128;
  float h[64];
  float ad[64];
  if (active) {
#pragma unroll
    for (int n = 0; n < 64; ++n) { h[n] = 0.0f; ad[n] = -__expf(Alog[(size_t)t * 64 + n]); }
  }
  for (int l = 0; l < L; ++l) {
    const int row = seq * L + l;
    if (t < 64)        Bs[t]      = dbl[(size_t)row * nx + dtr + t];
    else if (t < 128)  Cs[t - 64] = dbl[(size_t)row * nx + dtr + 64 + (t - 64)];
    __syncthreads();
    if (active) {
      float dt = dlt[(size_t)row * di + t];
      float u  = uc[(size_t)row * di + t];
      float du = dt * u;
      float y = 0.0f;
#pragma unroll
      for (int n = 0; n < 64; ++n) {
        h[n] = __expf(dt * ad[n]) * h[n] + du * Bs[n];
        y += h[n] * Cs[n];
      }
      y += u * Dp[t];
      float z = xz[(size_t)row * 2 * di + di + t];
      yv[(size_t)row * di + t] = y * silu_f(z);
    }
    __syncthreads();
  }
}

// ------------------------- spectral branch ---------------------------------
__global__ void spe_k(const float* __restrict__ x0,
                      const float* __restrict__ w7, const float* __restrict__ b7s,
                      const float* __restrict__ g7, const float* __restrict__ bb7,
                      const float* __restrict__ w3, const float* __restrict__ b3s,
                      const float* __restrict__ g3, const float* __restrict__ bb3,
                      float* __restrict__ s) {
  int row = blockIdx.x * blockDim.x + threadIdx.x;
  if (row >= 8192) return;
  const float* xr = x0 + (size_t)row * 64;
  float b7 = b7s[0], b3 = b3s[0];
  float s1 = 0.0f, q1 = 0.0f, s2 = 0.0f, q2 = 0.0f;
  for (int c = 0; c < 64; ++c) {
    float t1 = b7;
    for (int j = 0; j < 7; ++j) { int cc = c + j - 3; if (cc >= 0 && cc < 64) t1 += w7[j] * xr[cc]; }
    s1 += t1; q1 += t1 * t1;
    float t2 = b3;
    for (int j = 0; j < 3; ++j) { int cc = c + j - 1; if (cc >= 0 && cc < 64) t2 += w3[j] * xr[cc]; }
    s2 += t2; q2 += t2 * t2;
  }
  float m1 = s1 / 64.0f, r1 = rsqrtf(q1 / 64.0f - m1 * m1 + 1e-5f);
  float m2 = s2 / 64.0f, r2 = rsqrtf(q2 / 64.0f - m2 * m2 + 1e-5f);
  for (int c = 0; c < 64; ++c) {
    float t1 = b7;
    for (int j = 0; j < 7; ++j) { int cc = c + j - 3; if (cc >= 0 && cc < 64) t1 += w7[j] * xr[cc]; }
    float t2 = b3;
    for (int j = 0; j < 3; ++j) { int cc = c + j - 1; if (cc >= 0 && cc < 64) t2 += w3[j] * xr[cc]; }
    float u1 = gelu_f((t1 - m1) * r1 * g7[c] + bb7[c]);
    float u2 = gelu_f((t2 - m2) * r2 * g3[c] + bb3[c]);
    s[(size_t)row * 64 + c] = u1 + u2 + xr[c];
  }
}

__global__ void mean_hw_k(const float* __restrict__ s, float* __restrict__ xs0) {
  __shared__ float red[256];
  int bc = blockIdx.x;                         // 0..127 : b*64 + c
  int b = bc >> 6, c = bc & 63;
  float acc = 0.0f;
  for (int i = threadIdx.x; i < 4096; i += 256)
    acc += s[(((size_t)b * 4096 + i) * 64) + c];
  red[threadIdx.x] = acc;
  __syncthreads();
  for (int st = 128; st > 0; st >>= 1) {
    if (threadIdx.x < st) red[threadIdx.x] += red[threadIdx.x + st];
    __syncthreads();
  }
  if (threadIdx.x == 0) xs0[bc] = red[0] * (1.0f / 4096.0f);
}

__global__ void band_split_k(const float* __restrict__ xs0, float* __restrict__ xsb) {
  int i = blockIdx.x * blockDim.x + threadIdx.x;
  if (i >= 2 * 11 * 12) return;
  int k = i % 12, w = (i / 12) % 11, b = i / 132;
  int st = w * 6; if (st > 52) st = 52;
  xsb[i] = xs0[b * 64 + st + k];
}

// ------------------------- head --------------------------------------------
__global__ void head_k(const float* __restrict__ xspa, const float* __restrict__ xs,
                       const float* __restrict__ ow, const float* __restrict__ ob,
                       float* __restrict__ out) {
  int i = blockIdx.x * blockDim.x + threadIdx.x;
  if (i >= 2 * 16 * 4096) return;
  int hw = i & 4095, o = (i >> 12) & 15, b = i >> 16;
  const float* xr = xspa + ((size_t)b * 4096 + hw) * 64;
  const float* gr = xs + b * 64;
  const float* wr = ow + o * 64;
  float acc = ob[o];
#pragma unroll
  for (int c = 0; c < 64; ++c) acc += xr[c] * gr[c] * wr[c];
  out[i] = acc;
}

// ===========================================================================
// Host-side orchestration
// ===========================================================================
static inline int ceil_div(int a, int b) { return (a + b - 1) / b; }

static void gemm(hipStream_t st, const float* A, const float* W, const float* bias,
                 const float* bng, const float* bnb,
                 float* C, int M, int N, int K, int mode) {
  dim3 grid(ceil_div(M, 64), ceil_div(N, 64));
  gemm_wmma_k<<<grid, dim3(32, 4), 0, st>>>(A, W, bias, bng, bnb, C, M, N, K, mode);
}

struct MambaPtrs {
  const float *ln_g, *ln_b, *in_proj, *conv_w, *conv_b, *x_proj,
              *dt_w, *dt_b, *A_log, *D, *out_proj;
};
static MambaPtrs mk_mamba(void* const* d_in, int base) {
  MambaPtrs p;
  p.ln_g     = (const float*)d_in[base + 0];
  p.ln_b     = (const float*)d_in[base + 1];
  p.in_proj  = (const float*)d_in[base + 2];
  p.conv_w   = (const float*)d_in[base + 3];
  p.conv_b   = (const float*)d_in[base + 4];
  p.x_proj   = (const float*)d_in[base + 5];
  p.dt_w     = (const float*)d_in[base + 6];
  p.dt_b     = (const float*)d_in[base + 7];
  p.A_log    = (const float*)d_in[base + 8];
  p.D        = (const float*)d_in[base + 9];
  p.out_proj = (const float*)d_in[base + 10];
  return p;
}

// Residual(LayerNorm -> Mamba) in place on xio [Nseq*L, Dm].
static void mamba_block(hipStream_t st, float* xio, const MambaPtrs& p,
                        int Nseq, int L, int Dm, int di, int dtr,
                        float* xln, float* xz, float* uc, float* dbl,
                        float* dlt, float* yv) {
  const int NL = Nseq * L;
  ln_rows_k<<<ceil_div(NL, 256), 256, 0, st>>>(xio, xln, p.ln_g, p.ln_b, NL, Dm, 0);
  gemm(st, xln, p.in_proj, nullptr, nullptr, nullptr, xz, NL, 2 * di, Dm, GM_NONE);
  dwconv_silu_k<<<ceil_div(NL * di, 256), 256, 0, st>>>(xz, p.conv_w, p.conv_b, uc, Nseq, L, di);
  gemm(st, uc, p.x_proj, nullptr, nullptr, nullptr, dbl, NL, dtr + 128, di, GM_NONE);
  delta_k<<<ceil_div(NL * di, 256), 256, 0, st>>>(dbl, p.dt_w, p.dt_b, dlt, NL, di, dtr);
  scan_k<<<Nseq, 128, 0, st>>>(dbl, dlt, uc, xz, p.A_log, p.D, yv, L, di, dtr);
  gemm(st, yv, p.out_proj, nullptr, nullptr, nullptr, xio, NL, Dm, di, GM_ACC);
}

extern "C" void kernel_launch(void* const* d_in, const int* in_sizes, int n_in,
                              void* d_out, int out_size, void* d_ws, size_t ws_size,
                              hipStream_t stream) {
  (void)in_sizes; (void)n_in; (void)out_size; (void)ws_size;
  const float* X = (const float*)d_in[0];
#define CNN(i) ((const float*)d_in[(i)])
  // cnn params: indices 1..34 in dict order
  const float *pre_w = CNN(1),  *pre_b = CNN(2),  *pre_g = CNN(3),  *pre_bb = CNN(4);
  const float *chan_w = CNN(5), *chan_b = CNN(6), *chan_g = CNN(7), *chan_bb = CNN(8);
  const float *c1_w = CNN(9),   *c1_b = CNN(10),  *c1_g = CNN(11),  *c1_bb = CNN(12);
  const float *pool_g = CNN(13), *pool_bb = CNN(14);
  const float *fus_w = CNN(15), *fus_b = CNN(16);
  const float *nn1_w = CNN(17), *nn1_b = CNN(18), *nn1_g = CNN(19), *nn1_bb = CNN(20);
  const float *s1_w = CNN(21), *s1_b = CNN(22), *s1_g = CNN(23), *s1_bb = CNN(24);
  const float *s2_w = CNN(25), *s2_b = CNN(26), *s2_g = CNN(27), *s2_bb = CNN(28);
  const float *nn2_w = CNN(29), *nn2_b = CNN(30), *nn2_g = CNN(31), *nn2_bb = CNN(32);
  const float *out_w = CNN(33), *out_b = CNN(34);
  MambaPtrs local_spam = mk_mamba(d_in, 35);
  MambaPtrs spam       = mk_mamba(d_in, 46);
  MambaPtrs spem       = mk_mamba(d_in, 57);

  // ---- workspace carve-up (256B aligned) ----
  size_t off = 0;
  char* base = (char*)d_ws;
  auto alloc = [&](size_t elems) -> float* {
    float* p = (float*)(base + off);
    off += ((elems * sizeof(float) + 255) / 256) * 256;
    return p;
  };
  const int NTOK = 8192;                 // B*H*W
  const int NLmax = 12800;               // 50 windows * 256
  float* x0   = alloc((size_t)NTOK * 64);
  float* x1   = alloc((size_t)NTOK * 64);
  float* x2   = alloc((size_t)NTOK * 64);
  float* x3   = alloc((size_t)NTOK * 64);
  float* xcat = alloc((size_t)NTOK * 256);
  float* xf   = alloc((size_t)NTOK * 64);
  float* xw   = alloc((size_t)NLmax * 64);
  float* xa   = alloc((size_t)NTOK * 64);
  float* xspa = alloc((size_t)NTOK * 64);
  float* sspe = alloc((size_t)NTOK * 64);
  float* xs0  = alloc(128);
  float* xsb  = alloc(2 * 11 * 12);
  float* xs   = alloc(128);
  // mamba scratch (sized for the largest use: local windows)
  float* m_xln = alloc((size_t)NLmax * 64);
  float* m_xz  = alloc((size_t)NLmax * 256);
  float* m_uc  = alloc((size_t)NLmax * 128);
  float* m_dbl = alloc((size_t)NLmax * 132);
  float* m_dlt = alloc((size_t)NLmax * 128);
  float* m_yv  = alloc((size_t)NLmax * 128);

#define EW(n) ceil_div((n), 256), 256, 0, stream

  // ---- CNN trunk (GEMMs with fused BN/GELU epilogues) ----
  gemm(stream, X, pre_w, pre_b, pre_g, pre_bb, x0, NTOK, 64, 30, GM_BNGELU);
  gemm(stream, x0, chan_w, chan_b, chan_g, chan_bb, x1, NTOK, 64, 64, GM_BNGELU);
  conv3_k<<<EW(NTOK * 64)>>>(x0, c1_w, c1_b, c1_g, c1_bb, x2);
  pool5_k<<<EW(NTOK * 64)>>>(x0, pool_g, pool_bb, x3);
  concat_k<<<EW(NTOK * 256)>>>(x0, x1, x2, x3, xcat);
  gemm(stream, xcat, fus_w, fus_b, nullptr, nullptr, xf, NTOK, 64, 256, GM_GELU);

  // ---- local windowed Mamba (50 windows of 256 tokens) ----
  window_gather_k<<<EW(NLmax * 64)>>>(xf, xw);
  mamba_block(stream, xw, local_spam, 50, 256, 64, 128, 4,
              m_xln, m_xz, m_uc, m_dbl, m_dlt, m_yv);
  window_scatter_add_k<<<EW(NTOK * 64)>>>(xw, xf, xa);

  // ---- global spatial Mamba (2 sequences of 4096 tokens) ----
  mamba_block(stream, xa, spam, 2, 4096, 64, 128, 4,
              m_xln, m_xz, m_uc, m_dbl, m_dlt, m_yv);

  gemm(stream, xa, nn1_w, nn1_b, nullptr, nullptr, xspa, NTOK, 64, 64, GM_BIAS);
  ln_rows_k<<<EW(NTOK)>>>(xspa, xspa, nn1_g, nn1_bb, NTOK, 64, 1);

  // ---- spectral branch ----
  spe_k<<<EW(NTOK)>>>(x0, s1_w, s1_b, s1_g, s1_bb, s2_w, s2_b, s2_g, s2_bb, sspe);
  mean_hw_k<<<128, 256, 0, stream>>>(sspe, xs0);
  band_split_k<<<EW(2 * 11 * 12)>>>(xs0, xsb);
  mamba_block(stream, xsb, spem, 2, 11, 12, 24, 1,
              m_xln, m_xz, m_uc, m_dbl, m_dlt, m_yv);
  gemm(stream, xsb, nn2_w, nn2_b, nullptr, nullptr, xs, 2, 64, 132, GM_BIAS);
  ln_rows_k<<<EW(2)>>>(xs, xs, nn2_g, nn2_bb, 2, 64, 1);

  // ---- gated head -> logits [B,16,H,W] ----
  head_k<<<EW(2 * 16 * 4096)>>>(xspa, xs, out_w, out_b, (float*)d_out);
#undef EW
#undef CNN
}
